// MultiHeadAttentionPool_29334626632101
// MI455X (gfx1250) — compile-verified, matrix operand order checked by reading, not run on hardware
//
#include <hip/hip_runtime.h>
#include <hip/hip_bf16.h>

#define DIM 1024
#define HEAD_DIM 64
#define SEQ 4096
#define NB 16
#define NH 16
#define EPSLN 1e-5f

typedef __attribute__((ext_vector_type(16))) __bf16 v16bf;
typedef __attribute__((ext_vector_type(8)))  __bf16 v8bf;
typedef __attribute__((ext_vector_type(8)))  float  v8f;
typedef __attribute__((ext_vector_type(4)))  float  v4f;

union AFrag { v16bf v; v8bf h[2]; };

// ---------------------------------------------------------------------------
// 1) Convert w_k (fp32, row-major [n][k]) to bf16 once.
// ---------------------------------------------------------------------------
__global__ __launch_bounds__(256) void cvt_wk_kernel(const float* __restrict__ wk,
                                                     __bf16* __restrict__ wk16) {
    int i = blockIdx.x * 256 + threadIdx.x;
    wk16[i] = (__bf16)wk[i];
}

// ---------------------------------------------------------------------------
// 2) Normalize the query, fold layernorm weights:
//    qhat = LN(q); qw = qhat*w_norm; qws[h] = sum(qw); qb[h] = sum(qhat*b_norm)
// ---------------------------------------------------------------------------
__global__ void prepq_kernel(const float* __restrict__ query,
                             const float* __restrict__ w_norm,
                             const float* __restrict__ b_norm,
                             float* __restrict__ qw,
                             float* __restrict__ qws,
                             float* __restrict__ qb) {
    int h = blockIdx.x;
    int t = threadIdx.x;                    // 0..31
    float q0 = query[h * 64 + t];
    float q1 = query[h * 64 + t + 32];
    float s = q0 + q1;
    #pragma unroll
    for (int m = 1; m < 32; m <<= 1) s += __shfl_xor(s, m, 32);
    float mu = s * (1.0f / 64.0f);
    float d0 = q0 - mu, d1 = q1 - mu;
    float v = d0 * d0 + d1 * d1;
    #pragma unroll
    for (int m = 1; m < 32; m <<= 1) v += __shfl_xor(v, m, 32);
    float rs = rsqrtf(v * (1.0f / 64.0f) + EPSLN);
    float wn0 = w_norm[t], wn1 = w_norm[t + 32];
    float bn0 = b_norm[t], bn1 = b_norm[t + 32];
    float qh0 = d0 * rs * wn0 + bn0;
    float qh1 = d1 * rs * wn1 + bn1;
    qw[h * 64 + t]      = qh0 * wn0;
    qw[h * 64 + t + 32] = qh1 * wn1;
    float pw = qh0 * wn0 + qh1 * wn1;
    float pb = qh0 * bn0 + qh1 * bn1;
    #pragma unroll
    for (int m = 1; m < 32; m <<= 1) { pw += __shfl_xor(pw, m, 32); pb += __shfl_xor(pb, m, 32); }
    if (t == 0) { qws[h] = pw; qb[h] = pb; }
}

// ---------------------------------------------------------------------------
// 3) Fused K-projection (bf16 WMMA, f32 accum) + folded layernorm + score.
//    Grid: (ltile 0..63, nhalf 0..1, batch). Block 256 = 8 waves.
//    Workgroup tile: 64 rows x 512 cols. Wave w owns 64 rows x 64 cols
//    (= exactly one head): 4x4 WMMA tiles, each B fragment feeds 4 WMMAs.
//    B fragments are software-pipelined (ping-pong); next K-step's A data is
//    prefetched into registers during the WMMA phase.
// ---------------------------------------------------------------------------
__global__ __launch_bounds__(256) void kproj_scores_kernel(
    const float*  __restrict__ x,
    const __bf16* __restrict__ wk16,
    const float*  __restrict__ b_k,
    const float*  __restrict__ qw,
    const float*  __restrict__ qws,
    const float*  __restrict__ qb,
    float*        __restrict__ scores) {
    __shared__ __bf16 As[64][32];           // 64 rows x 32 K, bf16 (4 KB)

    const int ltile = blockIdx.x;
    const int nhalf = blockIdx.y;
    const int b     = blockIdx.z;
    const int tid    = threadIdx.x;
    const int wave   = tid >> 5;
    const int lane   = tid & 31;
    const int lane16 = lane & 15;
    const int half   = lane >> 4;
    const int nbase  = nhalf * 512 + wave * 64;
    const int head   = nhalf * 8 + wave;
    const int row0   = ltile * 64;

    v8f acc[4][4] = {};

    // A staging: thread -> (row r, 8 consecutive K), one ds_store_b128 per step.
    const int r  = tid >> 2;                // 0..63
    const int kk = (tid & 3) * 8;           // 0,8,16,24
    const float* xsrc = x + ((size_t)b * SEQ + row0 + r) * DIM + kk;
    // B: lane16 = w_k row (N column), half selects K blocks (mirrors A layout).
    const __bf16* wbase = wk16 + (size_t)(nbase + lane16) * DIM + half * 8;

    v4f xa0 = *(const v4f*)(xsrc);
    v4f xa1 = *(const v4f*)(xsrc + 4);

    for (int k0 = 0; k0 < DIM; k0 += 32) {
        __syncthreads();
        {
            v8bf pk;
            pk[0] = (__bf16)xa0.x; pk[1] = (__bf16)xa0.y;
            pk[2] = (__bf16)xa0.z; pk[3] = (__bf16)xa0.w;
            pk[4] = (__bf16)xa1.x; pk[5] = (__bf16)xa1.y;
            pk[6] = (__bf16)xa1.z; pk[7] = (__bf16)xa1.w;
            *(v8bf*)&As[r][kk] = pk;
        }
        __syncthreads();

        // Prefetch next K-step's A into registers (hidden behind WMMA phase).
        int kn = (k0 + 32 < DIM) ? (k0 + 32) : k0;
        xa0 = *(const v4f*)(xsrc + kn);
        xa1 = *(const v4f*)(xsrc + kn + 4);
        __builtin_prefetch(xsrc + ((k0 + 128 < DIM) ? (k0 + 128) : kn), 0, 3);

        // A fragments for the 4 M-subtiles.
        AFrag a[4];
        #pragma unroll
        for (int m = 0; m < 4; m++) {
            a[m].h[0] = *(const v8bf*)&As[m * 16 + lane16][half * 8];
            a[m].h[1] = *(const v8bf*)&As[m * 16 + lane16][16 + half * 8];
        }

        // B ping-pong pipeline: each fragment feeds 4 WMMAs.
        AFrag bq[2];
        bq[0].h[0] = *(const v8bf*)(wbase + k0);
        bq[0].h[1] = *(const v8bf*)(wbase + k0 + 16);
        #pragma unroll
        for (int nt = 0; nt < 4; nt++) {
            if (nt < 3) {
                const __bf16* p = wbase + (size_t)(nt + 1) * 16 * DIM + k0;
                bq[(nt + 1) & 1].h[0] = *(const v8bf*)(p);
                bq[(nt + 1) & 1].h[1] = *(const v8bf*)(p + 16);
            }
            #pragma unroll
            for (int m = 0; m < 4; m++)
                acc[m][nt] = __builtin_amdgcn_wmma_f32_16x16x32_bf16(
                    false, a[m].v, false, bq[nt & 1].v, (short)0, acc[m][nt],
                    false, false);
        }
    }

    // ---- Epilogue: folded layernorm + score. One head per wave. ----
    float qwv[4], bkv[4];
    #pragma unroll
    for (int t = 0; t < 4; t++) {
        bkv[t] = b_k[nbase + t * 16 + lane16];
        qwv[t] = qw[head * 64 + t * 16 + lane16];
    }
    const float qws_h = qws[head];
    const float qb_h  = qb[head];

    // C layout: VGPR j -> rows j (lanes 0-15) / j+8 (lanes 16-31), lanes = N.
    #pragma unroll
    for (int m = 0; m < 4; m++) {
        #pragma unroll
        for (int j = 0; j < 8; j++) {
            float s = 0.f, sq = 0.f, sw = 0.f;
            #pragma unroll
            for (int t = 0; t < 4; t++) {
                float c = acc[m][t][j] + bkv[t];
                s  += c;
                sq += c * c;
                sw += c * qwv[t];
            }
            #pragma unroll
            for (int msk = 1; msk < 16; msk <<= 1) {
                s  += __shfl_xor(s,  msk, 32);
                sq += __shfl_xor(sq, msk, 32);
                sw += __shfl_xor(sw, msk, 32);
            }
            if (lane16 == 0) {
                int row = row0 + m * 16 + j + half * 8;
                float mu  = s * (1.0f / 64.0f);
                float var = sq * (1.0f / 64.0f) - mu * mu;
                float rs  = rsqrtf(var + EPSLN);
                scores[(size_t)(b * NH + head) * SEQ + row] =
                    0.125f * (rs * (sw - mu * qws_h) + qb_h);
            }
        }
    }
}

// ---------------------------------------------------------------------------
// 4) Softmax over l (in place). One block per (b,h) row of 4096.
// ---------------------------------------------------------------------------
__global__ __launch_bounds__(256) void softmax_kernel(float* __restrict__ scores) {
    __shared__ float red[8];
    __shared__ float bcast;
    const size_t base = (size_t)blockIdx.x * SEQ;
    float* p = scores + base;
    int tid = threadIdx.x;

    float mx = -1e30f;
    for (int i = tid; i < SEQ; i += 256) mx = fmaxf(mx, p[i]);
    #pragma unroll
    for (int m = 1; m < 32; m <<= 1) mx = fmaxf(mx, __shfl_xor(mx, m, 32));
    if ((tid & 31) == 0) red[tid >> 5] = mx;
    __syncthreads();
    if (tid < 8) {
        float v = red[tid];
        #pragma unroll
        for (int m = 1; m < 8; m <<= 1) v = fmaxf(v, __shfl_xor(v, m, 32));
        if (tid == 0) bcast = v;
    }
    __syncthreads();
    mx = bcast;

    float sum = 0.f;
    for (int i = tid; i < SEQ; i += 256) {
        float e = __expf(p[i] - mx);
        p[i] = e;
        sum += e;
    }
    #pragma unroll
    for (int m = 1; m < 32; m <<= 1) sum += __shfl_xor(sum, m, 32);
    if ((tid & 31) == 0) red[tid >> 5] = sum;
    __syncthreads();
    if (tid < 8) {
        float v = red[tid];
        #pragma unroll
        for (int m = 1; m < 8; m <<= 1) v += __shfl_xor(v, m, 32);
        if (tid == 0) bcast = v;
    }
    __syncthreads();
    float inv = 1.0f / bcast;
    for (int i = tid; i < SEQ; i += 256) p[i] *= inv;
}

// ---------------------------------------------------------------------------
// 5) Attention-weighted pooling: xbar[b,h,:] += sum_l attn[b,h,l] * x[b,l,:]
// ---------------------------------------------------------------------------
__global__ __launch_bounds__(256) void pool_kernel(const float* __restrict__ x,
                                                   const float* __restrict__ attn,
                                                   float* __restrict__ xbar) {
    __shared__ float a_s[16][512];          // 32 KB
    const int jbase = blockIdx.x * 256;
    const int lbase = blockIdx.y * 512;
    const int b     = blockIdx.z;
    const int tid   = threadIdx.x;

    for (int i = tid; i < 16 * 512; i += 256) {
        int h = i >> 9, l = i & 511;
        a_s[h][l] = attn[(size_t)(b * NH + h) * SEQ + lbase + l];
    }
    __syncthreads();

    float accv[16] = {};
    const float* xp = x + ((size_t)b * SEQ + lbase) * DIM + jbase + tid;
    for (int l = 0; l < 512; l++) {
        float xv = xp[(size_t)l * DIM];
        #pragma unroll
        for (int h = 0; h < 16; h++) accv[h] += a_s[h][l] * xv;
    }
    #pragma unroll
    for (int h = 0; h < 16; h++)
        atomicAdd(&xbar[(size_t)(b * NH + h) * DIM + jbase + tid], accv[h]);
}

// ---------------------------------------------------------------------------
// 6) o[b,n] = xbar[b, n/64, :] . w_v[n,:] + b_v[n]   (V-proj collapsed)
// ---------------------------------------------------------------------------
__global__ __launch_bounds__(256) void vproj_kernel(const float* __restrict__ xbar,
                                                    const float* __restrict__ w_v,
                                                    const float* __restrict__ b_v,
                                                    float* __restrict__ o) {
    int n = blockIdx.x * 256 + threadIdx.x;
    int b = blockIdx.y;
    const float* xb = xbar + (size_t)(b * NH + (n >> 6)) * DIM;
    const float* w  = w_v + (size_t)n * DIM;
    float acc = 0.f;
    for (int j = 0; j < DIM; j += 4) {
        acc += xb[j] * w[j] + xb[j + 1] * w[j + 1] +
               xb[j + 2] * w[j + 2] + xb[j + 3] * w[j + 3];
    }
    o[b * DIM + n] = acc + b_v[n];
}

// ---------------------------------------------------------------------------
// 7) out[b,n] = o[b,:] . w_out[n,:] + b_out[n]
// ---------------------------------------------------------------------------
__global__ __launch_bounds__(256) void oproj_kernel(const float* __restrict__ o,
                                                    const float* __restrict__ w_out,
                                                    const float* __restrict__ b_out,
                                                    float* __restrict__ out) {
    int n = blockIdx.x * 256 + threadIdx.x;
    int b = blockIdx.y;
    const float* ob = o + (size_t)b * DIM;
    const float* w  = w_out + (size_t)n * DIM;
    float acc = 0.f;
    for (int j = 0; j < DIM; j += 4) {
        acc += ob[j] * w[j] + ob[j + 1] * w[j + 1] +
               ob[j + 2] * w[j + 2] + ob[j + 3] * w[j + 3];
    }
    out[b * DIM + n] = acc + b_out[n];
}

// ---------------------------------------------------------------------------
extern "C" void kernel_launch(void* const* d_in, const int* in_sizes, int n_in,
                              void* d_out, int out_size, void* d_ws, size_t ws_size,
                              hipStream_t stream) {
    const float* x      = (const float*)d_in[0];
    const float* query  = (const float*)d_in[1];
    const float* w_k    = (const float*)d_in[2];
    const float* b_k    = (const float*)d_in[3];
    const float* w_v    = (const float*)d_in[4];
    const float* b_v    = (const float*)d_in[5];
    const float* w_norm = (const float*)d_in[6];
    const float* b_norm = (const float*)d_in[7];
    const float* w_out  = (const float*)d_in[8];
    const float* b_out  = (const float*)d_in[9];
    float* out = (float*)d_out;

    char* ws = (char*)d_ws;
    const size_t OFF_QW     = (size_t)2 * 1024 * 1024;            // wk16: 2 MB
    const size_t OFF_QWS    = OFF_QW + 16 * 64 * sizeof(float);
    const size_t OFF_QB     = OFF_QWS + 256;
    const size_t OFF_SCORES = OFF_QB + 256;
    const size_t OFF_XBAR   = OFF_SCORES + (size_t)NB * NH * SEQ * sizeof(float);
    const size_t OFF_O      = OFF_XBAR + (size_t)NB * NH * DIM * sizeof(float);

    __bf16* wk16  = (__bf16*)ws;
    float* qw     = (float*)(ws + OFF_QW);
    float* qws    = (float*)(ws + OFF_QWS);
    float* qb     = (float*)(ws + OFF_QB);
    float* scores = (float*)(ws + OFF_SCORES);
    float* xbar   = (float*)(ws + OFF_XBAR);
    float* o      = (float*)(ws + OFF_O);

    hipMemsetAsync(xbar, 0, (size_t)NB * NH * DIM * sizeof(float), stream);
    cvt_wk_kernel<<<DIM * DIM / 256, 256, 0, stream>>>(w_k, wk16);
    prepq_kernel<<<NH, 32, 0, stream>>>(query, w_norm, b_norm, qw, qws, qb);
    kproj_scores_kernel<<<dim3(SEQ / 64, 2, NB), 256, 0, stream>>>(
        x, wk16, b_k, qw, qws, qb, scores);
    softmax_kernel<<<NB * NH, 256, 0, stream>>>(scores);
    pool_kernel<<<dim3(4, 8, NB), 256, 0, stream>>>(x, scores, xbar);
    vproj_kernel<<<dim3(4, NB), 256, 0, stream>>>(xbar, w_v, b_v, o);
    oproj_kernel<<<dim3(4, NB), 256, 0, stream>>>(o, w_out, b_out, out);
}
